// ConceptLayer_60868276519616
// MI455X (gfx1250) — matrix operand order repaired and checked by reading, not compiled, not run on hardware
//
#include <hip/hip_runtime.h>

typedef __attribute__((ext_vector_type(16))) _Float16 v16h;
typedef __attribute__((ext_vector_type(8)))  _Float16 v8h;
typedef __attribute__((ext_vector_type(8)))  float    v8f;

union V16 { v16h v; v8h h[2]; };

constexpr int kB = 8;
constexpr int kS = 2048;
constexpr int kE = 128;
constexpr int kTok = kB * kS;            // 16384
constexpr int kChunk = 256;
constexpr int kNChunk = kS / kChunk;     // 8
constexpr float kInvD = 1.0f / 1.2f;
constexpr float kLnEps = 1e-3f;

// ---------------------------------------------------------------------------
// Pre-swizzle dense_w (E x E fp32) into per-lane WMMA B-fragment order, f16.
// Fragment layout: [nt(8)][kc(4)][lane(32)][e(16)]
//   n = nt*16 + (lane&15);  k = kc*32 + (lane>>4)*8 + (e&7) + (e>>3)*16
// ---------------------------------------------------------------------------
__global__ __launch_bounds__(256) void prep_densew(const float* __restrict__ w,
                                                   _Float16* __restrict__ o) {
  int d = blockIdx.x * 256 + threadIdx.x;          // 16384 elements
  if (d >= 8 * 4 * 512) return;
  int e  = d & 15;
  int ln = (d >> 4) & 31;
  int kc = (d >> 9) & 3;
  int nt = d >> 11;
  int n = nt * 16 + (ln & 15);
  int k = kc * 32 + (ln >> 4) * 8 + (e & 7) + (e >> 3) * 16;
  o[d] = (_Float16)w[k * kE + n];
}

// ---------------------------------------------------------------------------
// Pre-swizzle concept_map (E x E x E fp32) into B-fragment order, f16.
// Fragment layout: [ct(8)][a(128)][pc(4)][lane(32)][e(16)]
//   c = ct*16 + (lane&15);  p = pc*32 + (lane>>4)*8 + (e&7) + (e>>3)*16
// ---------------------------------------------------------------------------
__global__ __launch_bounds__(256) void prep_concept(const float* __restrict__ w,
                                                    _Float16* __restrict__ o) {
  int d = blockIdx.x * 256 + threadIdx.x;          // 2,097,152 elements
  int e  = d & 15;
  int ln = (d >> 4) & 31;
  int pc = (d >> 9) & 3;
  int a  = (d >> 11) & 127;
  int ct = d >> 18;
  int c = ct * 16 + (ln & 15);
  int p = pc * 32 + (ln >> 4) * 8 + (e & 7) + (e >> 3) * 16;
  o[d] = (_Float16)w[((long)c * kE + a) * kE + p];
}

// ---------------------------------------------------------------------------
// Dense: h = x @ W + b.  32 tokens/block, 8 waves (one 16-col N-tile each).
// Writes h in f32 (scan + residual) and f16 (A-fragment source).
// ---------------------------------------------------------------------------
__global__ __launch_bounds__(256) void dense_wmma(const float* __restrict__ x,
                                                  const _Float16* __restrict__ wdsw,
                                                  const float* __restrict__ bias,
                                                  float* __restrict__ h32,
                                                  _Float16* __restrict__ h16) {
  __shared__ alignas(16) _Float16 xl[32 * 136];
  int tid  = threadIdx.x;
  int tok0 = blockIdx.x * 32;
  for (int i = tid; i < 32 * 128; i += 256) {
    int t = i >> 7, k = i & 127;
    xl[t * 136 + k] = (_Float16)x[(long)(tok0 + t) * kE + k];
  }
  __syncthreads();

  int wave = tid >> 5, lane = tid & 31;
  int m = lane & 15, hf = lane >> 4;
  v8f acc0 = {}, acc1 = {};
  const _Float16* wb = wdsw + wave * 4 * 512 + lane * 16;

#pragma unroll
  for (int kc = 0; kc < 4; ++kc) {
    V16 Bf;
    Bf.h[0] = *(const v8h*)(wb + kc * 512);
    Bf.h[1] = *(const v8h*)(wb + kc * 512 + 8);
    int base = kc * 32 + hf * 8;
    V16 A0, A1;
    A0.h[0] = *(const v8h*)(xl + m * 136 + base);
    A0.h[1] = *(const v8h*)(xl + m * 136 + base + 16);
    A1.h[0] = *(const v8h*)(xl + (m + 16) * 136 + base);
    A1.h[1] = *(const v8h*)(xl + (m + 16) * 136 + base + 16);
    acc0 = __builtin_amdgcn_wmma_f32_16x16x32_f16(false, A0.v, false, Bf.v,
                                                  (short)0, acc0, false, false);
    acc1 = __builtin_amdgcn_wmma_f32_16x16x32_f16(false, A1.v, false, Bf.v,
                                                  (short)0, acc1, false, false);
  }

  int c = wave * 16 + (lane & 15);
  float bv = bias[c];
#pragma unroll
  for (int r = 0; r < 8; ++r) {
    int t0 = r + hf * 8;                      // D row M = r + 8*half
    float v0 = acc0[r] + bv;
    float v1 = acc1[r] + bv;
    long i0 = (long)(tok0 + t0) * kE + c;
    long i1 = (long)(tok0 + 16 + t0) * kE + c;
    h32[i0] = v0; h16[i0] = (_Float16)v0;
    h32[i1] = v1; h16[i1] = (_Float16)v1;
  }
}

// ---------------------------------------------------------------------------
// Scan pass 1: per-chunk local recurrence l_k (l_0=0, l_k=(l_{k-1}+h)/d),
// store l_k and the chunk total T = l_L.  64 blocks x 128 threads.
// ---------------------------------------------------------------------------
__global__ __launch_bounds__(128) void scan_local(const float* __restrict__ h32,
                                                  float* __restrict__ lscr,
                                                  float* __restrict__ Tc) {
  int e = threadIdx.x;
  int blk = blockIdx.x;                     // b*kNChunk + chunk
  int b = blk >> 3, c = blk & 7;
  long base = ((long)b * kS + c * kChunk) * kE + e;
  float carry = 0.f;
  for (int k = 0; k < kChunk; k += 8) {
    float hv[8];
#pragma unroll
    for (int j = 0; j < 8; ++j) hv[j] = h32[base + (long)(k + j) * kE];
#pragma unroll
    for (int j = 0; j < 8; ++j) {
      lscr[base + (long)(k + j) * kE] = carry;
      carry = (carry + hv[j]) * kInvD;
    }
  }
  Tc[blk * kE + e] = carry;
}

// ---------------------------------------------------------------------------
// Scan pass 2: chunk carries sigma_{c+1} = sigma_c * d^-L + T_c.
// ---------------------------------------------------------------------------
__global__ __launch_bounds__(128) void scan_carry(const float* __restrict__ Tc,
                                                  float* __restrict__ Sig) {
  int e = threadIdx.x;
  int b = blockIdx.x;
  float powl = 1.f;
#pragma unroll 1
  for (int i = 0; i < kChunk; ++i) powl *= kInvD;   // 1.2^-256 ~ 5.4e-21 (f32 ok)
  float sig = 0.f;
  for (int c = 0; c < kNChunk; ++c) {
    Sig[(b * kNChunk + c) * kE + e] = sig;
    sig = sig * powl + Tc[(b * kNChunk + c) * kE + e];
  }
}

// ---------------------------------------------------------------------------
// Scan pass 3: s_t = l_k + sigma * d^-k, written as f16 for the WMMA A build.
// ---------------------------------------------------------------------------
__global__ __launch_bounds__(128) void scan_finish(const float* __restrict__ lscr,
                                                   const float* __restrict__ Sig,
                                                   _Float16* __restrict__ s16) {
  int e = threadIdx.x;
  int blk = blockIdx.x;
  int b = blk >> 3, c = blk & 7;
  long base = ((long)b * kS + c * kChunk) * kE + e;
  float sig = Sig[blk * kE + e];
  float scale = 1.f;
  for (int k = 0; k < kChunk; k += 8) {
    float lv[8];
#pragma unroll
    for (int j = 0; j < 8; ++j) lv[j] = lscr[base + (long)(k + j) * kE];
#pragma unroll
    for (int j = 0; j < 8; ++j) {
      s16[base + (long)(k + j) * kE] = (_Float16)(lv[j] + sig * scale);
      scale *= kInvD;
    }
  }
}

// ---------------------------------------------------------------------------
// Fused contraction + residual + LayerNorm.
// out[t,c] = sum_{a,p} h[t,a] s[t,p] W[c,a,p]  == GEMM M=16384,N=128,K=16384
// A rows are outer products built on the fly: A[t, a*128+p] = h16[t,a]*s16[t,p].
// Per block: 32 tokens; 8 waves each own one 16-col c-tile; K loop = 128 a x 4 p-chunks.
// s-fragments are K-invariant in 'a' -> preloaded into registers once.
// ---------------------------------------------------------------------------
__global__ __launch_bounds__(256) void concept_wmma(const _Float16* __restrict__ h16g,
                                                    const _Float16* __restrict__ s16g,
                                                    const float* __restrict__ h32g,
                                                    const _Float16* __restrict__ wsw,
                                                    const float* __restrict__ gam,
                                                    const float* __restrict__ bet,
                                                    float* __restrict__ out) {
  __shared__ alignas(16) _Float16 sl[32 * 136];   // s16 tile, padded
  __shared__ alignas(16) _Float16 hT[128 * 32];   // h16 tile, transposed [a][t]
  __shared__ alignas(16) float rbuf[32 * 132];    // residual buffer, padded
  int tid  = threadIdx.x;
  int tok0 = blockIdx.x * 32;

  for (int i = tid; i < 32 * 128; i += 256) {
    int t = i >> 7, p = i & 127;
    sl[t * 136 + p] = s16g[(long)(tok0 + t) * kE + p];
    hT[p * 32 + t]  = h16g[(long)(tok0 + t) * kE + p];
  }
  __syncthreads();

  int wave = tid >> 5, lane = tid & 31;
  int m = lane & 15, hf = lane >> 4;

  // Preload s fragments (reused by all 128 a-blocks): 2 M-tiles x 4 p-chunks.
  V16 sf[2][4];
#pragma unroll
  for (int mt = 0; mt < 2; ++mt)
#pragma unroll
    for (int pc = 0; pc < 4; ++pc) {
      const _Float16* sp = sl + (mt * 16 + m) * 136 + pc * 32 + hf * 8;
      sf[mt][pc].h[0] = *(const v8h*)sp;
      sf[mt][pc].h[1] = *(const v8h*)(sp + 16);
    }

  v8f acc0 = {}, acc1 = {};
  const _Float16* wbase = wsw + (long)wave * 128 * 2048 + lane * 16;

#pragma unroll 2
  for (int a = 0; a < 128; ++a) {
    _Float16 h0 = hT[a * 32 + m];
    _Float16 h1 = hT[a * 32 + 16 + m];
    const _Float16* wa = wbase + a * 2048;
    __builtin_prefetch(wa + 2048, 0, 0);
#pragma unroll
    for (int pc = 0; pc < 4; ++pc) {
      V16 Bf;
      Bf.h[0] = *(const v8h*)(wa + pc * 512);
      Bf.h[1] = *(const v8h*)(wa + pc * 512 + 8);
      v16h A0 = sf[0][pc].v * h0;       // 8x v_pk_mul_f16
      v16h A1 = sf[1][pc].v * h1;
      acc0 = __builtin_amdgcn_wmma_f32_16x16x32_f16(false, A0, false, Bf.v,
                                                    (short)0, acc0, false, false);
      acc1 = __builtin_amdgcn_wmma_f32_16x16x32_f16(false, A1, false, Bf.v,
                                                    (short)0, acc1, false, false);
    }
  }

  // Residual add into LDS
  int c = wave * 16 + (lane & 15);
#pragma unroll
  for (int r = 0; r < 8; ++r) {
    int t0 = r + hf * 8;
    rbuf[t0 * 132 + c]        = acc0[r] + h32g[(long)(tok0 + t0) * kE + c];
    rbuf[(16 + t0) * 132 + c] = acc1[r] + h32g[(long)(tok0 + 16 + t0) * kE + c];
  }
  __syncthreads();

  // LayerNorm: each wave owns 4 tokens; 8 lanes/token, 16 channels/lane.
  int t  = wave * 4 + (lane >> 3);
  int c0 = (lane & 7) * 16;
  float sum = 0.f, sq = 0.f;
#pragma unroll
  for (int j = 0; j < 16; ++j) {
    float v = rbuf[t * 132 + c0 + j];
    sum += v; sq += v * v;
  }
  sum += __shfl_xor(sum, 1); sq += __shfl_xor(sq, 1);
  sum += __shfl_xor(sum, 2); sq += __shfl_xor(sq, 2);
  sum += __shfl_xor(sum, 4); sq += __shfl_xor(sq, 4);
  float mean = sum * (1.f / 128.f);
  float var  = sq * (1.f / 128.f) - mean * mean;
  float inv  = rsqrtf(var + kLnEps);
  long ob = (long)(tok0 + t) * kE;
#pragma unroll
  for (int j = 0; j < 16; ++j) {
    float v = rbuf[t * 132 + c0 + j];
    out[ob + c0 + j] = (v - mean) * inv * gam[c0 + j] + bet[c0 + j];
  }
}

// ---------------------------------------------------------------------------
extern "C" void kernel_launch(void* const* d_in, const int* in_sizes, int n_in,
                              void* d_out, int out_size, void* d_ws, size_t ws_size,
                              hipStream_t stream) {
  (void)in_sizes; (void)n_in; (void)out_size; (void)ws_size;
  const float* x   = (const float*)d_in[0];
  const float* dw  = (const float*)d_in[1];
  const float* db  = (const float*)d_in[2];
  const float* cm  = (const float*)d_in[3];
  const float* gam = (const float*)d_in[4];
  const float* bet = (const float*)d_in[5];
  float* out = (float*)d_out;

  char* ws = (char*)d_ws;
  float*    H32  = (float*)   (ws + 0);          //  8 MB  h fp32
  _Float16* HS16 = (_Float16*)(ws + 8388608);    //  4 MB  h f16
  _Float16* S16  = (_Float16*)(ws + 12582912);   //  4 MB  s f16
  float*    LSCR = (float*)   (ws + 16777216);   //  8 MB  local-scan scratch
  float*    TC   = (float*)   (ws + 25165824);   // 32 KB  chunk totals
  float*    SIG  = (float*)   (ws + 25198592);   // 32 KB  chunk carries
  _Float16* WSW  = (_Float16*)(ws + 25231360);   //  4 MB  swizzled concept_map
  _Float16* WDSW = (_Float16*)(ws + 29425664);   // 32 KB  swizzled dense_w

  prep_densew <<<64,   256, 0, stream>>>(dw, WDSW);
  prep_concept<<<8192, 256, 0, stream>>>(cm, WSW);
  dense_wmma  <<<kTok / 32, 256, 0, stream>>>(x, WDSW, db, H32, HS16);
  scan_local  <<<kB * kNChunk, 128, 0, stream>>>(H32, LSCR, TC);
  scan_carry  <<<kB, 128, 0, stream>>>(TC, SIG);
  scan_finish <<<kB * kNChunk, 128, 0, stream>>>(LSCR, SIG, S16);
  concept_wmma<<<kTok / 32, 256, 0, stream>>>(HS16, S16, H32, WSW, gam, bet, out);
}
